// SNNModel_45492293599317
// MI455X (gfx1250) — matrix operand order repaired and checked by reading, not compiled, not run on hardware
//
#include <hip/hip_runtime.h>
#include <hip/hip_bf16.h>
#include <stdint.h>

// Problem constants (from reference)
#define T_STEPS 100
#define BATCH   256
#define N_IN    1024
#define N_HID   4096
#define N_OUT   512
#define BETA    0.95f
#define THRESH  1.0f

typedef __bf16 bf16_t;
typedef __attribute__((ext_vector_type(16))) __bf16 v16bf;
typedef __attribute__((ext_vector_type(8)))  float  v8f;

// Types required by the async-to-LDS builtins (from hipcc diagnostics):
//   b128 takes 'int __attribute__((vector_size(16)))*', b64 takes vector_size(8).
typedef int v4i_ld __attribute__((__vector_size__(16)));
typedef int v2i_ld __attribute__((__vector_size__(8)));

// ---------------- CDNA5 async-to-LDS helpers (guarded; fallback = plain copy) --
#if defined(__gfx1250__) && \
    __has_builtin(__builtin_amdgcn_global_load_async_to_lds_b128) && \
    __has_builtin(__builtin_amdgcn_global_load_async_to_lds_b64)
#define USE_ASYNC 1
#else
#define USE_ASYNC 0
#endif

__device__ __forceinline__ void async_cp_b128(void* lds, const void* gsrc) {
#if USE_ASYNC
  __builtin_amdgcn_global_load_async_to_lds_b128((v4i_ld*)gsrc, (v4i_ld*)lds, 0, 0);
#else
  *(uint4*)lds = *(const uint4*)gsrc;
#endif
}

__device__ __forceinline__ void async_cp_b64(void* lds, const void* gsrc) {
#if USE_ASYNC
  __builtin_amdgcn_global_load_async_to_lds_b64((v2i_ld*)gsrc, (v2i_ld*)lds, 0, 0);
#else
  *(uint2*)lds = *(const uint2*)gsrc;
#endif
}

template<int N>
__device__ __forceinline__ void wait_async() {
#if USE_ASYNC
#if __has_builtin(__builtin_amdgcn_s_wait_asynccnt)
  __builtin_amdgcn_s_wait_asynccnt(N);
#else
  asm volatile("s_wait_asynccnt %0" :: "n"(N));
#endif
#endif
}

// ---------------- fused GEMM (bf16 WMMA) + LIF neuron update ------------------
// C[m,n] = sum_k A[m,k] * W[n,k] + bias[n];  then LIF:
//   rst = (mem_old > TH) ? TH : 0; mem = BETA*mem_old + C - rst; spk = mem > TH
// Layer1 (LAST=false): A = x_t (f32, converted while staging), spikes -> bf16 ws
// Layer2 (LAST=true):  A = spk1 (bf16 ws), spikes+mem -> f32 d_out slices
template<int TM, int TN, int K, int N, bool A_F32, bool LAST>
__launch_bounds__(256)
__global__ void gemm_lif_kernel(const void* __restrict__ Aptr,
                                const bf16_t* __restrict__ Wbf,
                                const float* __restrict__ bias,
                                float* __restrict__ mem,
                                bf16_t* __restrict__ spk_bf,
                                float* __restrict__ spk_out,
                                float* __restrict__ mem_out) {
#if defined(__gfx1250__)
  constexpr int KT   = 32;                 // K per WMMA step (bf16: 16x16x32)
  constexpr int LDK  = KT + 8;             // padded LDS row: 80B = 20-bank stride
  constexpr int WROWS = 2, WCOLS = 4;      // 8 wave32s per workgroup
  constexpr int WM = TM / WROWS, WN = TN / WCOLS;
  constexpr int FM = WM / 16, FN = WN / 16;
  constexpr int NK = K / KT;
  constexpr int NB  = (TN * 4) / 256;              // 16B B-chunks per thread
  constexpr int NA8 = A_F32 ? 0 : (TM / 32);       // 8B  A-chunks per thread (bf16 A)
  constexpr int NAP = A_F32 ? (TM / 16) : 0;       // float2 A-pairs per thread
  constexpr int NASYNC = NB + NA8;                 // async ops issued per stage

  __shared__ __align__(16) bf16_t Asm[2][TM][LDK];
  __shared__ __align__(16) bf16_t Bsm[2][TN][LDK];

  const int tid   = threadIdx.x;
  const int lane  = tid & 31;
  const int wave  = tid >> 5;
  const int wrow  = wave / WCOLS;
  const int wcol  = wave % WCOLS;
  const int tileN = blockIdx.x;
  const int tileM = blockIdx.y;
  const int half  = lane >> 4;    // wave32 half (ISA frag layouts)
  const int l16   = lane & 15;

  const float*  Af  = (const float*)Aptr;
  const bf16_t* Abf = (const bf16_t*)Aptr;

  v8f acc[FM][FN] = {};

  // ---- stage one K-slab (KT cols) of A and B tiles into LDS buffer `buf` ----
  auto stage = [&](int buf, int kk) {
    const int k0 = kk * KT;
    // B tile: TN x 32 bf16 from W[n][k] row-major (already B^T layout). 4x16B/row.
#pragma unroll
    for (int i = 0; i < NB; ++i) {
      const int c   = tid + i * 256;
      const int row = c >> 2;
      const int off = (c & 3) * 8;
      async_cp_b128(&Bsm[buf][row][off],
                    Wbf + (size_t)(tileN * TN + row) * K + k0 + off);
    }
    if constexpr (A_F32) {
      // A tile: load f32 x, convert to bf16 pairs while storing to LDS.
#pragma unroll
      for (int i = 0; i < NAP; ++i) {
        const int p   = tid + i * 256;
        const int row = p >> 4;
        const int col = (p & 15) * 2;
        const float2 v =
            *(const float2*)(Af + (size_t)(tileM * TM + row) * K + k0 + col);
        union { bf16_t h[2]; uint32_t u; } pk;
        pk.h[0] = (bf16_t)v.x;
        pk.h[1] = (bf16_t)v.y;
        *(uint32_t*)&Asm[buf][row][col] = pk.u;
      }
    } else {
      // A tile: bf16 spikes, straight async copy. 8x8B per row.
#pragma unroll
      for (int i = 0; i < NA8; ++i) {
        const int c   = tid + i * 256;
        const int row = c >> 3;
        const int off = (c & 7) * 4;
        async_cp_b64(&Asm[buf][row][off],
                     Abf + (size_t)(tileM * TM + row) * K + k0 + off);
      }
    }
  };

  // ---- consume one staged slab: FM*FN wmma_f32_16x16x32_bf16 ----------------
  auto compute = [&](int buf) {
    union Frag { uint4 q[2]; v16bf v; };
    Frag a[FM], b[FN];
#pragma unroll
    for (int fm = 0; fm < FM; ++fm) {
      // A frag (16x32 bf16): lane m = l16; k = [half*8,+8) and [16+half*8,+8)
      const bf16_t* arow = &Asm[buf][wrow * WM + fm * 16 + l16][0];
      a[fm].q[0] = *(const uint4*)(arow + half * 8);
      a[fm].q[1] = *(const uint4*)(arow + 16 + half * 8);
    }
#pragma unroll
    for (int fn = 0; fn < FN; ++fn) {
      // B frag (32x16 bf16): lane n = l16; k = [half*8,+8) and [16+half*8,+8)
      const bf16_t* brow = &Bsm[buf][wcol * WN + fn * 16 + l16][0];
      b[fn].q[0] = *(const uint4*)(brow + half * 8);
      b[fn].q[1] = *(const uint4*)(brow + 16 + half * 8);
    }
#pragma unroll
    for (int fm = 0; fm < FM; ++fm) {
#pragma unroll
      for (int fn = 0; fn < FN; ++fn) {
        acc[fm][fn] = __builtin_amdgcn_wmma_f32_16x16x32_bf16(
            false, a[fm].v, false, b[fn].v, (short)0, acc[fm][fn], false, false);
      }
    }
  };

  // ---- double-buffered main loop over K --------------------------------------
  stage(0, 0);
  int buf = 0;
  for (int kk = 0; kk < NK; ++kk) {
    if (kk + 1 < NK) {
      stage(buf ^ 1, kk + 1);     // prefetch next slab into other buffer
      wait_async<NASYNC>();       // current slab's async ops (older) complete
    } else {
      wait_async<0>();
    }
    __syncthreads();
    compute(buf);
    __syncthreads();              // everyone done reading before restage
    buf ^= 1;
  }

  // ---- fused LIF epilogue ----------------------------------------------------
  // C layout: VGPR r -> (m = r + 8*half, n = l16) within each 16x16 frag.
#pragma unroll
  for (int fm = 0; fm < FM; ++fm) {
#pragma unroll
    for (int fn = 0; fn < FN; ++fn) {
      const int ng = tileN * TN + wcol * WN + fn * 16 + l16;
      const float bn = bias[ng];
#pragma unroll
      for (int r = 0; r < 8; ++r) {
        const int mg = tileM * TM + wrow * WM + fm * 16 + half * 8 + r;
        const size_t idx = (size_t)mg * N + ng;
        const float cur = acc[fm][fn][r] + bn;
        const float mo  = mem[idx];
        const float rst = (mo > THRESH) ? THRESH : 0.0f;   // detached reset
        const float mn  = BETA * mo + cur - rst;
        mem[idx] = mn;
        const float s = (mn > THRESH) ? 1.0f : 0.0f;
        if constexpr (LAST) {
          spk_out[idx] = s;
          mem_out[idx] = mn;
        } else {
          spk_bf[idx] = (bf16_t)s;   // {0,1} exact in bf16
        }
      }
    }
  }
#endif // __gfx1250__
}

// ---------------- tiny helper kernels ----------------------------------------
__global__ void cvt_f32_bf16_kernel(const float* __restrict__ src,
                                    bf16_t* __restrict__ dst, int n) {
  const int i = blockIdx.x * blockDim.x + threadIdx.x;
  if (i < n) dst[i] = (bf16_t)src[i];
}

__global__ void zero_f32_kernel(float* __restrict__ p, int n) {
  const int i = blockIdx.x * blockDim.x + threadIdx.x;
  if (i < n) p[i] = 0.0f;
}

// ---------------- launch ------------------------------------------------------
extern "C" void kernel_launch(void* const* d_in, const int* in_sizes, int n_in,
                              void* d_out, int out_size, void* d_ws, size_t ws_size,
                              hipStream_t stream) {
  (void)in_sizes; (void)n_in; (void)out_size; (void)ws_size;
  const float* x  = (const float*)d_in[0];   // [T,B,N_IN]
  const float* W1 = (const float*)d_in[1];   // [N_HID,N_IN]
  const float* b1 = (const float*)d_in[2];   // [N_HID]
  const float* W2 = (const float*)d_in[3];   // [N_OUT,N_HID]
  const float* b2 = (const float*)d_in[4];   // [N_OUT]
  float* out = (float*)d_out;                // [spk_rec | mem_rec], each [T,B,N_OUT]

  // workspace carve-out (~19 MB total)
  char* ws = (char*)d_ws;
  size_t off = 0;
  auto carve = [&](size_t bytes) -> char* {
    char* p = ws + off;
    off += (bytes + 255) & ~(size_t)255;
    return p;
  };
  bf16_t* W1bf = (bf16_t*)carve((size_t)N_HID * N_IN * sizeof(bf16_t));   // 8 MB
  bf16_t* W2bf = (bf16_t*)carve((size_t)N_OUT * N_HID * sizeof(bf16_t));  // 4 MB
  bf16_t* spk1 = (bf16_t*)carve((size_t)BATCH * N_HID * sizeof(bf16_t));  // 2 MB
  float*  mem1 = (float*)carve((size_t)BATCH * N_HID * sizeof(float));    // 4 MB
  float*  mem2 = (float*)carve((size_t)BATCH * N_OUT * sizeof(float));    // 0.5 MB

  // weights -> bf16 once; membrane state -> 0 (must re-init every call)
  cvt_f32_bf16_kernel<<<(N_HID * N_IN + 255) / 256, 256, 0, stream>>>(
      W1, W1bf, N_HID * N_IN);
  cvt_f32_bf16_kernel<<<(N_OUT * N_HID + 255) / 256, 256, 0, stream>>>(
      W2, W2bf, N_OUT * N_HID);
  zero_f32_kernel<<<(BATCH * N_HID + 255) / 256, 256, 0, stream>>>(
      mem1, BATCH * N_HID);
  zero_f32_kernel<<<(BATCH * N_OUT + 255) / 256, 256, 0, stream>>>(
      mem2, BATCH * N_OUT);

  const size_t outT = (size_t)T_STEPS * BATCH * N_OUT;
  for (int t = 0; t < T_STEPS; ++t) {
    // layer 1: [256,1024] x [1024,4096] -> spk1 (bf16)
    gemm_lif_kernel<64, 128, N_IN, N_HID, true, false>
        <<<dim3(N_HID / 128, BATCH / 64), 256, 0, stream>>>(
            x + (size_t)t * BATCH * N_IN, W1bf, b1, mem1, spk1,
            nullptr, nullptr);
    // layer 2: [256,4096] x [4096,512] -> spk_rec[t], mem_rec[t] (f32)
    gemm_lif_kernel<32, 64, N_HID, N_OUT, false, true>
        <<<dim3(N_OUT / 64, BATCH / 32), 256, 0, stream>>>(
            spk1, W2bf, b2, mem2, nullptr,
            out + (size_t)t * BATCH * N_OUT,
            out + outT + (size_t)t * BATCH * N_OUT);
  }
}
